// CrossAttentionGraphBlock_72335839199365
// MI455X (gfx1250) — matrix operand order repaired and checked by reading, not compiled, not run on hardware
//
#include <hip/hip_runtime.h>
#include <math.h>

// ---------------- problem constants (from reference) ----------------
#define BB   8
#define SS   1024
#define EE   512
#define FF   256
#define HH   8
#define DD   64

typedef __attribute__((ext_vector_type(16))) _Float16 v16h;
typedef __attribute__((ext_vector_type(8)))  _Float16 v8h;
typedef __attribute__((ext_vector_type(8)))  float    v8f;
typedef __attribute__((ext_vector_type(4)))  int      i32x4;

// ---- CDNA5 async global->LDS staging (guarded; falls back to plain copies) ----
#if defined(__has_builtin)
#if __has_builtin(__builtin_amdgcn_global_load_async_to_lds_b128)
#define HAVE_ASYNC 1
#endif
#endif

static __device__ __forceinline__ void cp16(void* lds, const void* g) {
#ifdef HAVE_ASYNC
    // signature (from compiler diagnostic): (i32x4 AS1*, i32x4 AS3*, imm offset, imm cpol)
    __builtin_amdgcn_global_load_async_to_lds_b128(
        (__attribute__((address_space(1))) i32x4*)g,
        (__attribute__((address_space(3))) i32x4*)lds, 0, 0);
#else
    *(v8h*)lds = *(const v8h*)g;
#endif
}
static __device__ __forceinline__ void tiles_join() {
#ifdef HAVE_ASYNC
    asm volatile("s_wait_asynccnt 0" ::: "memory");
#endif
    __syncthreads();
}

static __device__ __forceinline__ v8f wmma_f16(v16h a, v16h b, v8f c) {
    return __builtin_amdgcn_wmma_f32_16x16x32_f16(
        false, a, false, b, (short)0, c, false, false);
}

union FragU { v16h v; v8h h[2]; };

// =====================================================================
// 64(M) x 128(N) tile GEMM, all-f16 operands, f32 accumulate.
//   C = A[M,K](f16) @ Wt[N,K](f16, pre-transposed) + bias
// grid = (N/128, M/64), block = 256 (8 waves, 4 accum tiles each).
// Cf (f32) and/or Ch (f16) outputs; fuse==1: leakyReLU + residual(f32).
// =====================================================================
__global__ __launch_bounds__(256) void gemm_h(
    const _Float16* __restrict__ A, int lda,
    const _Float16* __restrict__ Wt, int K,
    const float* __restrict__ bias,
    float* __restrict__ Cf, _Float16* __restrict__ Ch, int ldc,
    const float* __restrict__ residual, int fuse)
{
    __shared__ _Float16 As[64][40];    // [m][k], 80B rows (16B aligned)
    __shared__ _Float16 Bs[128][40];   // [n][k]

    const int tid  = threadIdx.x;
    const int wave = tid >> 5, lane = tid & 31;
    const int hi   = lane >> 4, lm = lane & 15;
    const int m0   = blockIdx.y * 64;
    const int n0   = blockIdx.x * 128;
    const int wm   = (wave & 3) * 16;
    const int wn   = (wave >> 2) * 64;

    v8f c[4] = {};

    const int ar = tid >> 2, ac = (tid & 3) * 8;    // A: 64 x 32, 8 halfs/thr
    const int br = tid >> 1, bc = (tid & 1) * 16;   // B: 128 x 32, 16 halfs/thr

    for (int k0 = 0; k0 < K; k0 += 32) {
        cp16(&As[ar][ac],     A  + (size_t)(m0 + ar) * lda + k0 + ac);
        cp16(&Bs[br][bc],     Wt + (size_t)(n0 + br) * K   + k0 + bc);
        cp16(&Bs[br][bc + 8], Wt + (size_t)(n0 + br) * K   + k0 + bc + 8);
        tiles_join();

        FragU a;
        a.h[0] = *(const v8h*)&As[wm + lm][hi * 16];
        a.h[1] = *(const v8h*)&As[wm + lm][hi * 16 + 8];
        #pragma unroll
        for (int j = 0; j < 4; ++j) {
            FragU b;
            b.h[0] = *(const v8h*)&Bs[wn + j * 16 + lm][hi * 16];
            b.h[1] = *(const v8h*)&Bs[wn + j * 16 + lm][hi * 16 + 8];
            c[j] = wmma_f16(a.v, b.v, c[j]);
        }
        __syncthreads();
    }

    #pragma unroll
    for (int j = 0; j < 4; ++j) {
        #pragma unroll
        for (int r = 0; r < 8; ++r) {
            const int row = m0 + wm + r + 8 * hi;
            const int col = n0 + wn + j * 16 + lm;
            float v = c[j][r] + bias[col];
            if (fuse == 1)
                v = (v > 0.f ? v : 0.01f * v) + residual[(size_t)row * ldc + col];
            if (Cf) Cf[(size_t)row * ldc + col] = v;
            if (Ch) Ch[(size_t)row * ldc + col] = (_Float16)v;
        }
    }
}

// =====================================================================
// Flash attention, f16 operands. grid = (S/64, H, B), block = 256.
// 1/sqrt(D) folded into the score write.
// =====================================================================
__global__ __launch_bounds__(256) void attn64_wmma(
    const _Float16* __restrict__ qi, const _Float16* __restrict__ ki,
    const _Float16* __restrict__ vi, const int* __restrict__ counts,
    _Float16* __restrict__ ctxh)
{
    __shared__ _Float16 Qs[64][72];   // [q][d]
    __shared__ _Float16 Ks[64][72];   // [kk][d]
    __shared__ _Float16 Vs[64][72];   // [d][kk] (transposed)
    __shared__ _Float16 Ps[64][72];   // [q][kk] probs
    __shared__ float    Sb[64][65];
    __shared__ float    rowM[64], rowL[64], corr[64];

    const int tid  = threadIdx.x;
    const int wave = tid >> 5, lane = tid & 31;
    const int hi   = lane >> 4, lm = lane & 15;
    const int q0   = blockIdx.x * 64;
    const int h    = blockIdx.y;
    const int b    = blockIdx.z;
    const size_t base = (size_t)b * SS * EE + (size_t)h * DD;

    const int lr = tid >> 2, lc = (tid & 3) * 16;  // 64 rows x 4 chunks of 16

    {   // raw async copy of Q tile
        const _Float16* qp = qi + base + (size_t)(q0 + lr) * EE + lc;
        cp16(&Qs[lr][lc], qp);
        cp16(&Qs[lr][lc + 8], qp + 8);
    }
    if (tid < 64) { rowM[tid] = -1e30f; rowL[tid] = 0.f; }

    const int wm = (wave & 3) * 16;
    const int wn = (wave >> 2) * 32;
    v8f o0 = {}, o1 = {};

    const int cnt = counts[b];
    const int nkt = (cnt + 63) >> 6;
    tiles_join();

    for (int kt = 0; kt < nkt; ++kt) {
        const int kb = kt * 64;
        {   // K: raw async copy; V: manual transpose
            const _Float16* kp = ki + base + (size_t)(kb + lr) * EE + lc;
            const _Float16* vp = vi + base + (size_t)(kb + lr) * EE + lc;
            cp16(&Ks[lr][lc], kp);
            cp16(&Ks[lr][lc + 8], kp + 8);
            v8h va = *(const v8h*)vp;
            v8h vb = *(const v8h*)(vp + 8);
            #pragma unroll
            for (int i = 0; i < 8; ++i) {
                Vs[lc + i][lr]     = va[i];
                Vs[lc + 8 + i][lr] = vb[i];
            }
        }
        tiles_join();

        v8f s0 = {}, s1 = {};
        #pragma unroll
        for (int kk = 0; kk < 64; kk += 32) {
            FragU a, b0, b1;
            a.h[0]  = *(const v8h*)&Qs[wm + lm][kk + hi * 16];
            a.h[1]  = *(const v8h*)&Qs[wm + lm][kk + hi * 16 + 8];
            b0.h[0] = *(const v8h*)&Ks[wn + lm][kk + hi * 16];
            b0.h[1] = *(const v8h*)&Ks[wn + lm][kk + hi * 16 + 8];
            b1.h[0] = *(const v8h*)&Ks[wn + 16 + lm][kk + hi * 16];
            b1.h[1] = *(const v8h*)&Ks[wn + 16 + lm][kk + hi * 16 + 8];
            s0 = wmma_f16(a.v, b0.v, s0);
            s1 = wmma_f16(a.v, b1.v, s1);
        }
        #pragma unroll
        for (int r = 0; r < 8; ++r) {
            const int m = wm + r + 8 * hi;
            const int n0c = wn + lm, n1c = n0c + 16;
            Sb[m][n0c] = (kb + n0c < cnt) ? s0[r] * 0.125f : -1e30f;
            Sb[m][n1c] = (kb + n1c < cnt) ? s1[r] * 0.125f : -1e30f;
        }
        __syncthreads();

        if (tid < 64) {     // online softmax, one thread per query row
            float mx = rowM[tid], tm = -1e30f;
            for (int j = 0; j < 64; ++j) tm = fmaxf(tm, Sb[tid][j]);
            const float mn = fmaxf(mx, tm);
            const float cf = __expf(mx - mn);
            float sum = 0.f;
            for (int j = 0; j < 64; ++j) {
                const float p = __expf(Sb[tid][j] - mn);
                Ps[tid][j] = (_Float16)p;
                sum += p;
            }
            rowL[tid] = rowL[tid] * cf + sum;
            rowM[tid] = mn;
            corr[tid] = cf;
        }
        __syncthreads();

        #pragma unroll
        for (int r = 0; r < 8; ++r) {
            const float cf = corr[wm + r + 8 * hi];
            o0[r] *= cf; o1[r] *= cf;
        }
        #pragma unroll
        for (int kk = 0; kk < 64; kk += 32) {
            FragU a, b0, b1;
            a.h[0]  = *(const v8h*)&Ps[wm + lm][kk + hi * 16];
            a.h[1]  = *(const v8h*)&Ps[wm + lm][kk + hi * 16 + 8];
            b0.h[0] = *(const v8h*)&Vs[wn + lm][kk + hi * 16];
            b0.h[1] = *(const v8h*)&Vs[wn + lm][kk + hi * 16 + 8];
            b1.h[0] = *(const v8h*)&Vs[wn + 16 + lm][kk + hi * 16];
            b1.h[1] = *(const v8h*)&Vs[wn + 16 + lm][kk + hi * 16 + 8];
            o0 = wmma_f16(a.v, b0.v, o0);
            o1 = wmma_f16(a.v, b1.v, o1);
        }
        __syncthreads();
    }

    #pragma unroll
    for (int r = 0; r < 8; ++r) {
        const int m = wm + r + 8 * hi;
        const float inv = 1.f / rowL[m];
        _Float16* cp = ctxh + base + (size_t)(q0 + m) * EE;
        cp[wn + lm]      = (_Float16)(o0[r] * inv);
        cp[wn + 16 + lm] = (_Float16)(o1[r] * inv);
    }
}

// =====================================================================
// LayerNorm over E=512 (block per row); optional residual, dual outputs.
// =====================================================================
__global__ __launch_bounds__(256) void ln512(
    const float* __restrict__ a, const float* __restrict__ res,
    const float* __restrict__ g, const float* __restrict__ bta,
    float* __restrict__ out, _Float16* __restrict__ outh)
{
    __shared__ float red[256];
    const int row = blockIdx.x, tid = threadIdx.x;
    const size_t o = (size_t)row * EE;
    float x0 = a[o + tid], x1 = a[o + tid + 256];
    if (res) { x0 += res[o + tid]; x1 += res[o + tid + 256]; }

    red[tid] = x0 + x1;
    __syncthreads();
    for (int s = 128; s > 0; s >>= 1) { if (tid < s) red[tid] += red[tid + s]; __syncthreads(); }
    const float mean = red[0] * (1.f / EE);
    __syncthreads();

    const float d0 = x0 - mean, d1 = x1 - mean;
    red[tid] = d0 * d0 + d1 * d1;
    __syncthreads();
    for (int s = 128; s > 0; s >>= 1) { if (tid < s) red[tid] += red[tid + s]; __syncthreads(); }
    const float rstd = rsqrtf(red[0] * (1.f / EE) + 1e-5f);

    const float r0 = d0 * rstd * g[tid]       + bta[tid];
    const float r1 = d1 * rstd * g[tid + 256] + bta[tid + 256];
    if (out)  { out[o + tid] = r0;             out[o + tid + 256] = r1; }
    if (outh) { outh[o + tid] = (_Float16)r0;  outh[o + tid + 256] = (_Float16)r1; }
}

// ---------------- prep kernels ----------------
__global__ void zero_f32(float* __restrict__ p, int n) {
    int i = blockIdx.x * 256 + threadIdx.x;
    if (i < n) p[i] = 0.f;
}

__global__ void cvt_h(const float* __restrict__ a, _Float16* __restrict__ o, int n) {
    int i = blockIdx.x * 256 + threadIdx.x;
    if (i < n) o[i] = (_Float16)a[i];
}

// Wt[n][k] = (f16) W[k][col0 + n]   (one-time weight transpose+convert)
__global__ void wtransp(const float* __restrict__ W, int ldw, int col0,
                        _Float16* __restrict__ Wt, int N, int K) {
    int idx = blockIdx.x * 256 + threadIdx.x;
    if (idx >= N * K) return;
    const int n = idx / K, k2 = idx - n * K;
    Wt[idx] = (_Float16)W[(size_t)k2 * ldw + col0 + n];
}

__global__ void count_starts(const int* __restrict__ bids, int total,
                             int* __restrict__ counts, int* __restrict__ starts) {
    __shared__ int c[BB];
    const int tid = threadIdx.x;
    if (tid < BB) c[tid] = 0;
    __syncthreads();
    for (int i = tid; i < total; i += 256) atomicAdd(&c[bids[i]], 1);
    __syncthreads();
    if (tid == 0) {
        int run = 0;
        for (int b = 0; b < BB; ++b) { counts[b] = c[b]; starts[b] = run; run += c[b]; }
    }
}

__global__ void scatter_nodes(const float* __restrict__ nodes,
                              const int* __restrict__ bids,
                              const int* __restrict__ starts,
                              _Float16* __restrict__ padded, int total) {
    int idx = blockIdx.x * 256 + threadIdx.x;
    if (idx >= total * FF) return;
    const int i = idx / FF, f = idx - i * FF;
    const int b = bids[i];
    const int pos = i - starts[b];
    padded[((size_t)b * SS + pos) * FF + f] = (_Float16)nodes[idx];
}

// =====================================================================
extern "C" void kernel_launch(void* const* d_in, const int* in_sizes, int n_in,
                              void* d_out, int out_size, void* d_ws, size_t ws_size,
                              hipStream_t stream) {
    const float* x    = (const float*)d_in[0];
    const float* nodes= (const float*)d_in[1];
    const float* wq   = (const float*)d_in[2];
    const float* bq   = (const float*)d_in[3];
    const float* wk   = (const float*)d_in[4];
    const float* bk   = (const float*)d_in[5];
    const float* wv   = (const float*)d_in[6];
    const float* bv   = (const float*)d_in[7];
    const float* in_w = (const float*)d_in[8];
    const float* in_b = (const float*)d_in[9];
    const float* wo   = (const float*)d_in[10];
    const float* bo   = (const float*)d_in[11];
    const float* g1w  = (const float*)d_in[12];
    const float* b1w  = (const float*)d_in[13];
    const float* w1   = (const float*)d_in[14];
    const float* bd1  = (const float*)d_in[15];
    const float* g2w  = (const float*)d_in[16];
    const float* b2w  = (const float*)d_in[17];
    const int*   bids = (const int*)d_in[18];
    const int    total = in_sizes[18];

    const size_t NP   = (size_t)BB * SS * FF;  // padded elements (f16)
    const size_t NBSE = (size_t)BB * SS * EE;  // activation elements

    float* ws = (float*)d_ws;
    size_t off = 0;                                  // offsets in floats
    _Float16* padded = (_Float16*)(ws + off); off += NP / 2;
    _Float16* xh     = (_Float16*)(ws + off); off += NBSE / 2;
    _Float16* qh     = (_Float16*)(ws + off); off += NBSE / 2;
    _Float16* kh     = (_Float16*)(ws + off); off += NBSE / 2;
    _Float16* vh     = (_Float16*)(ws + off); off += NBSE / 2;
    _Float16* qih    = (_Float16*)(ws + off); off += NBSE / 2;
    _Float16* kih    = (_Float16*)(ws + off); off += NBSE / 2;
    _Float16* vih    = (_Float16*)(ws + off); off += NBSE / 2;
    float*    attn_o = ws + off;              off += NBSE;
    float*    hbuf   = ws + off;              off += NBSE;
    _Float16* wqT    = (_Float16*)(ws + off); off += (EE * EE) / 2;
    _Float16* wkT    = (_Float16*)(ws + off); off += (EE * EE) / 2;
    _Float16* wvT    = (_Float16*)(ws + off); off += (EE * FF) / 2;
    _Float16* wiqT   = (_Float16*)(ws + off); off += (EE * EE) / 2;
    _Float16* wikT   = (_Float16*)(ws + off); off += (EE * EE) / 2;
    _Float16* wivT   = (_Float16*)(ws + off); off += (EE * EE) / 2;
    _Float16* woT    = (_Float16*)(ws + off); off += (EE * EE) / 2;
    _Float16* w1T    = (_Float16*)(ws + off); off += (EE * EE) / 2;
    int* counts = (int*)(ws + off);
    int* starts = counts + BB;
    // buffer reuse (dead after their consumer GEMMs)
    _Float16* ctxh = qh;
    _Float16* hh   = kh;
    float*    ybuf = attn_o;
    float*    outp = (float*)d_out;

    // ---- unbatch + pad (f16) ----
    zero_f32<<<(int)((NP / 2 + 255) / 256), 256, 0, stream>>>((float*)padded, (int)(NP / 2));
    count_starts<<<1, 256, 0, stream>>>(bids, total, counts, starts);
    scatter_nodes<<<(total * FF + 255) / 256, 256, 0, stream>>>(nodes, bids, starts, padded, total);
    // ---- one-time conversions ----
    cvt_h<<<(int)((NBSE + 255) / 256), 256, 0, stream>>>(x, xh, (int)NBSE);
    wtransp<<<(EE * EE + 255) / 256, 256, 0, stream>>>(wq,   EE,     0,      wqT,  EE, EE);
    wtransp<<<(EE * EE + 255) / 256, 256, 0, stream>>>(wk,   EE,     0,      wkT,  EE, EE);
    wtransp<<<(EE * FF + 255) / 256, 256, 0, stream>>>(wv,   EE,     0,      wvT,  EE, FF);
    wtransp<<<(EE * EE + 255) / 256, 256, 0, stream>>>(in_w, 3 * EE, 0,      wiqT, EE, EE);
    wtransp<<<(EE * EE + 255) / 256, 256, 0, stream>>>(in_w, 3 * EE, EE,     wikT, EE, EE);
    wtransp<<<(EE * EE + 255) / 256, 256, 0, stream>>>(in_w, 3 * EE, 2 * EE, wivT, EE, EE);
    wtransp<<<(EE * EE + 255) / 256, 256, 0, stream>>>(wo,   EE,     0,      woT,  EE, EE);
    wtransp<<<(EE * EE + 255) / 256, 256, 0, stream>>>(w1,   EE,     0,      w1T,  EE, EE);

    const dim3 gg(EE / 128, (BB * SS) / 64);   // (4, 128)
    // ---- external projections (f16 out) ----
    gemm_h<<<gg, 256, 0, stream>>>(xh,     EE, wqT,  EE, bq,          nullptr, qh,  EE, nullptr, 0);
    gemm_h<<<gg, 256, 0, stream>>>(xh,     EE, wkT,  EE, bk,          nullptr, kh,  EE, nullptr, 0);
    gemm_h<<<gg, 256, 0, stream>>>(padded, FF, wvT,  FF, bv,          nullptr, vh,  EE, nullptr, 0);
    // ---- MHA in-proj (f16 out) ----
    gemm_h<<<gg, 256, 0, stream>>>(qh,     EE, wiqT, EE, in_b,        nullptr, qih, EE, nullptr, 0);
    gemm_h<<<gg, 256, 0, stream>>>(kh,     EE, wikT, EE, in_b + EE,   nullptr, kih, EE, nullptr, 0);
    gemm_h<<<gg, 256, 0, stream>>>(vh,     EE, wivT, EE, in_b + 2*EE, nullptr, vih, EE, nullptr, 0);
    // ---- attention ----
    const dim3 ga(SS / 64, HH, BB);
    attn64_wmma<<<ga, 256, 0, stream>>>(qih, kih, vih, counts, ctxh);
    // ---- out-proj (f32 out for LN/residual) ----
    gemm_h<<<gg, 256, 0, stream>>>(ctxh, EE, woT, EE, bo, attn_o, nullptr, EE, nullptr, 0);
    // ---- h = LN(attn_out + x), dual precision ----
    ln512<<<BB * SS, 256, 0, stream>>>(attn_o, x, g1w, b1w, hbuf, hh);
    // ---- y = leaky(h @ w1 + bd1) + h (fused, f32 out; reuses attn_o) ----
    gemm_h<<<gg, 256, 0, stream>>>(hh, EE, w1T, EE, bd1, ybuf, nullptr, EE, hbuf, 1);
    // ---- out = LN(y) ----
    ln512<<<BB * SS, 256, 0, stream>>>(ybuf, nullptr, g2w, b2w, outp, nullptr);
}
